// MultiHeadAttention_51273319579772
// MI455X (gfx1250) — matrix operand order repaired and checked by reading, not compiled, not run on hardware
//
#include <hip/hip_runtime.h>

typedef __attribute__((ext_vector_type(16))) __bf16 v16bf;
typedef __attribute__((ext_vector_type(8)))  __bf16 v8bf;
typedef __attribute__((ext_vector_type(8)))  float  v8f;
typedef __attribute__((ext_vector_type(4)))  float  v4f;
typedef __attribute__((ext_vector_type(4)))  unsigned int v4u;
typedef __attribute__((ext_vector_type(8)))  int    v8i;
typedef __attribute__((ext_vector_type(4)))  int    v4i;

static constexpr int Bd = 4, Sd = 2048, Dd = 1024, Hd = 16, HDd = 64;
static constexpr int Md = Bd * Sd;   // 8192 rows

#define WMMA_BF16(a, b, c) \
  __builtin_amdgcn_wmma_f32_16x16x32_bf16(false, (a), false, (b), (short)0, (c), false, false)

// ---------------------------------------------------------------------------
// Fragment loaders (CDNA5 wave32 WMMA register layouts, 16x16x32 bf16)
// A-matrix 16x32: lanes 0-15 hold K {0..7,16..23}; lanes 16-31 hold {8..15,24..31}
// B-matrix 32x16: lanes 0-15 hold K {0..7,16..23}; lanes 16-31 hold {8..15,24..31}
// ---------------------------------------------------------------------------
__device__ __forceinline__ v16bf load_afrag_bf16(const __bf16* __restrict__ row,
                                                 int kbase, bool hi) {
  const int o = kbase + (hi ? 8 : 0);
  v16bf a;
  *(v8bf*)&a         = *(const v8bf*)(row + o);
  *(((v8bf*)&a) + 1) = *(const v8bf*)(row + o + 16);
  return a;
}

__device__ __forceinline__ v16bf load_bfrag_bf16(const __bf16* __restrict__ col,
                                                 bool hi) {
  const __bf16* q = col + (hi ? 8 : 0);
  v16bf a;
  *(v8bf*)&a         = *(const v8bf*)q;
  *(((v8bf*)&a) + 1) = *(const v8bf*)(q + 16);
  return a;
}

// ---------------------------------------------------------------------------
// Tensor Data Mover: async 2D tile load (rows x 32 bf16) global -> LDS.
// D# per CDNA5 ISA ch.8: group0 = {count=1 | lds_addr | global_addr | type=2},
// group1 = {data_size=2B, pad 16B per 64B row, tensor dims/strides, tile dims}.
// LDS row stride becomes 64+16 = 80 bytes (conflict-spreading padding).
// ---------------------------------------------------------------------------
__device__ __forceinline__ void tdm_load_tile_a(const __bf16* gsrc, unsigned ldsOff,
                                                unsigned K, unsigned Mrows) {
  const unsigned long long ga = (unsigned long long)gsrc;
  v4u g0 = { 1u,                                            // count=1, user D#
             ldsOff,                                        // LDS byte address
             (unsigned)(ga & 0xffffffffu),                  // global_addr[31:0]
             (unsigned)((ga >> 32) & 0x01ffffffu) | 0x80000000u };  // [56:32] | type=2
  // dword0: data_size=1(2B)<<16 | pad_enable<<20 | pad_interval=3(16 DW)<<22
  //         | pad_amount=3(4 DW)<<25
  v8i g1 = { (int)((1u << 16) | (1u << 20) | (3u << 22) | (3u << 25)),
             (int)(K << 16),          // tensor_dim0[15:0] in [31:16]
             (int)(Mrows << 16),      // tensor_dim0[31:16]=0, tensor_dim1[15:0]
             (int)(32u << 16),        // tensor_dim1[31:16]=0, tile_dim0=32
             (int)128,                // tile_dim1=128 rows, tile_dim2=0
             (int)K,                  // tensor_dim0_stride[31:0]
             0, 0 };                  // stride hi, tensor_dim1_stride
  v4i z4 = { 0, 0, 0, 0 };
#if __clang_major__ >= 23
  v8i z8 = { 0, 0, 0, 0, 0, 0, 0, 0 };
  __builtin_amdgcn_tensor_load_to_lds(g0, g1, z4, z4, z8, 0);
#else
  __builtin_amdgcn_tensor_load_to_lds(g0, g1, z4, z4, 0);
#endif
}

// ---------------------------------------------------------------------------
// Prep 1: elementwise fp32 -> bf16 (X). 8 elements/thread, b128 in / b128 out.
// ---------------------------------------------------------------------------
__global__ __launch_bounds__(256) void convert_bf16(const float* __restrict__ in,
                                                    __bf16* __restrict__ out) {
  const size_t i = ((size_t)blockIdx.x * 256 + threadIdx.x) * 8;
  v4f x0 = *(const v4f*)(in + i);
  v4f x1 = *(const v4f*)(in + i + 4);
  v8bf o;
#pragma unroll
  for (int e = 0; e < 4; ++e) {
    o[e]     = (__bf16)x0[e];
    o[4 + e] = (__bf16)x1[e];
  }
  *(v8bf*)(out + i) = o;
}

// ---------------------------------------------------------------------------
// Prep 2: W[K][N] fp32 -> Wt[N][K] bf16, 32x32 LDS-tiled transpose.
// ---------------------------------------------------------------------------
__global__ __launch_bounds__(256) void transpose_cvt(const float* __restrict__ W,
                                                     __bf16* __restrict__ Wt,
                                                     int K, int N) {
  __shared__ __bf16 tile[32][33];
  const int tx = threadIdx.x & 31;
  const int ty = threadIdx.x >> 5;  // 0..7
  const int n0 = blockIdx.x * 32;
  const int k0 = blockIdx.y * 32;
#pragma unroll
  for (int j = 0; j < 4; ++j) {
    const int k = ty + j * 8;
    tile[tx][k] = (__bf16)W[(size_t)(k0 + k) * N + n0 + tx];  // coalesced along n
  }
  __syncthreads();
#pragma unroll
  for (int j = 0; j < 4; ++j) {
    const int n = ty + j * 8;
    Wt[(size_t)(n0 + n) * K + k0 + tx] = tile[n][tx];  // coalesced along k
  }
}

// ---------------------------------------------------------------------------
// GEMM: out = A[M,K](bf16) x Wt[N,K](bf16, pre-transposed) + bias.
// A block tile (128x32) is staged to LDS by the Tensor Data Mover, double-
// buffered (issue wave 0, TENSORcnt wait + one barrier per K-step) so the DMA
// overlaps the WMMA stream and A is read from global once per block instead of
// once per N-wave. B-fragments are direct global loads, two 16B per lane (weights
// are 2 MB bf16, L2-resident). 8 waves in 2(M) x 4(N); wave tile 64x64
// (16 WMMA per K-step); block tile 128x256, BK=32.
// OUT_MODE 0: bf16 head-split [B,H,S,HD]        (Q, K)
// OUT_MODE 1: bf16 head-split transposed [B,H,HD,S]  (V)
// OUT_MODE 2: f32 row-major [M,N]               (final projection)
// ---------------------------------------------------------------------------
template <int OUT_MODE>
__global__ __launch_bounds__(256) void gemm_wmma(const __bf16* __restrict__ A,
                                                 const __bf16* __restrict__ Wt,
                                                 const float* __restrict__ bias,
                                                 void* __restrict__ out,
                                                 int K, int N) {
  // 2 buffers x 128 rows x 40 elems (32 data + 8 pad) = 20.5 KB
  __shared__ __bf16 abuf[2][128 * 40];

  const int t = threadIdx.x;
  const int wave = t >> 5;
  const int lane = t & 31;
  const int ln = lane & 15;
  const bool hi = lane >= 16;
  const int mwave = wave >> 2;  // 0..1
  const int nwave = wave & 3;   // 0..3
  const int rowBase = blockIdx.x * 128 + mwave * 64;
  const int colBase = blockIdx.y * 256 + nwave * 64;

  const __bf16* Ablk = A + (size_t)blockIdx.x * 128 * K;  // block's A rows
  const __bf16* brow = Wt + (size_t)(colBase + ln) * K;

  const unsigned lds0 = (unsigned)(unsigned long long)(void*)&abuf[0][0];
  const unsigned lds1 = (unsigned)(unsigned long long)(void*)&abuf[1][0];

  if (wave == 0) tdm_load_tile_a(Ablk, lds0, K, Md);  // prologue: buf0 <- k0=0

  v8f acc[4][4] = {};
  int cur = 0;

  for (int k0 = 0; k0 < K; k0 += 32) {
    __builtin_amdgcn_s_wait_tensorcnt(0);  // no-op for waves that issued nothing
    __syncthreads();                       // publish abuf[cur] to all waves
    if (wave == 0 && k0 + 32 < K)
      tdm_load_tile_a(Ablk + (k0 + 32), cur ? lds0 : lds1, K, Md);

    v16bf bfr[4];
#pragma unroll
    for (int j = 0; j < 4; ++j)
      bfr[j] = load_bfrag_bf16(brow + (size_t)j * 16 * K + k0, hi);
#pragma unroll
    for (int i = 0; i < 4; ++i) {
      const __bf16* arow = &abuf[cur][(mwave * 64 + i * 16 + ln) * 40];
      v16bf afr = load_afrag_bf16(arow, 0, hi);
#pragma unroll
      for (int j = 0; j < 4; ++j)
        acc[i][j] = WMMA_BF16(afr, bfr[j], acc[i][j]);
    }
    cur ^= 1;
  }

  // Epilogue: C layout -> VGPR r holds row (r | hi*8), N = lane%16.
#pragma unroll
  for (int i = 0; i < 4; ++i) {
#pragma unroll
    for (int j = 0; j < 4; ++j) {
      const int col = colBase + j * 16 + ln;
      const float bv = bias[col];
#pragma unroll
      for (int r = 0; r < 8; ++r) {
        const int row = rowBase + i * 16 + r + (hi ? 8 : 0);
        const float v = acc[i][j][r] + bv;
        if constexpr (OUT_MODE == 2) {
          ((float*)out)[(size_t)row * N + col] = v;
        } else {
          const int b = row / Sd, s = row % Sd;
          const int h = col / HDd, hd = col % HDd;
          __bf16* o = (__bf16*)out;
          if constexpr (OUT_MODE == 0)
            o[(((size_t)b * Hd + h) * Sd + s) * HDd + hd] = (__bf16)v;
          else
            o[(((size_t)b * Hd + h) * HDd + hd) * Sd + s] = (__bf16)v;
        }
      }
    }
  }
}

// ---------------------------------------------------------------------------
// Flash attention: one block per (64-query tile, head, batch); 4 waves.
// Q,K in [B,H,S,HD] bf16; V transposed [B,H,HD,S] bf16.
// scores = QK^T * 1/sqrt(64) + mask; online softmax; ctx = P V.
// ---------------------------------------------------------------------------
__global__ __launch_bounds__(128) void flash_attn(const __bf16* __restrict__ Q,
                                                  const __bf16* __restrict__ Kh,
                                                  const __bf16* __restrict__ Vt,
                                                  const float* __restrict__ mask,
                                                  __bf16* __restrict__ ctx_out) {
  __shared__ __bf16 p_lds[4][16][64];  // per-wave P staging (C-layout -> A-layout)

  const int t = threadIdx.x;
  const int wave = t >> 5;
  const int lane = t & 31;
  const int ln = lane & 15;
  const bool hi = lane >= 16;

  const int qTile = blockIdx.x;
  const int h = blockIdx.y;
  const int b = blockIdx.z;
  const size_t bh = (size_t)b * Hd + h;
  const __bf16* Qb = Q + bh * Sd * HDd;
  const __bf16* Kb = Kh + bh * Sd * HDd;
  const __bf16* Vb = Vt + bh * HDd * Sd;  // [hd][s]
  const float* mb = mask + (size_t)b * Sd;

  const int qRow = qTile * 64 + wave * 16 + ln;
  v16bf qf0 = load_afrag_bf16(Qb + (size_t)qRow * HDd, 0, hi);
  v16bf qf1 = load_afrag_bf16(Qb + (size_t)qRow * HDd, 32, hi);

  v8f ctx[4] = {};
  float m_i[8], l_i[8];
#pragma unroll
  for (int r = 0; r < 8; ++r) { m_i[r] = -1e30f; l_i[r] = 0.0f; }

  for (int kb = 0; kb < Sd; kb += 64) {
    // ---- scores = Q K^T ----
    v8f sc[4] = {};
#pragma unroll
    for (int kt = 0; kt < 4; ++kt) {
      const __bf16* krow = Kb + (size_t)(kb + kt * 16 + ln) * HDd;
      v16bf b0 = load_bfrag_bf16(krow, hi);       // hd 0..31
      v16bf b1 = load_bfrag_bf16(krow + 32, hi);  // hd 32..63
      sc[kt] = WMMA_BF16(qf0, b0, sc[kt]);
      sc[kt] = WMMA_BF16(qf1, b1, sc[kt]);
    }
    // scale + additive mask (broadcast over queries)
#pragma unroll
    for (int kt = 0; kt < 4; ++kt) {
      const float mval = mb[kb + kt * 16 + ln];
#pragma unroll
      for (int r = 0; r < 8; ++r) sc[kt][r] = sc[kt][r] * 0.125f + mval;
    }
    // ---- online softmax: row max across 4 key-tiles + 16 lanes ----
#pragma unroll
    for (int r = 0; r < 8; ++r) {
      float v = fmaxf(fmaxf(sc[0][r], sc[1][r]), fmaxf(sc[2][r], sc[3][r]));
      v = fmaxf(v, __shfl_xor(v, 1, 32));
      v = fmaxf(v, __shfl_xor(v, 2, 32));
      v = fmaxf(v, __shfl_xor(v, 4, 32));
      v = fmaxf(v, __shfl_xor(v, 8, 32));
      const float mnew = fmaxf(m_i[r], v);
      const float rs = __expf(m_i[r] - mnew);
      m_i[r] = mnew;
      l_i[r] *= rs;
#pragma unroll
      for (int ht = 0; ht < 4; ++ht) ctx[ht][r] *= rs;
    }
    __syncthreads();
    // ---- p = exp(s - m); stash bf16 into LDS in A-matrix-friendly layout ----
#pragma unroll
    for (int kt = 0; kt < 4; ++kt) {
#pragma unroll
      for (int r = 0; r < 8; ++r) {
        const float p = __expf(sc[kt][r] - m_i[r]);
        sc[kt][r] = p;
        p_lds[wave][r + (hi ? 8 : 0)][kt * 16 + ln] = (__bf16)p;
      }
    }
#pragma unroll
    for (int r = 0; r < 8; ++r) {
      float s = sc[0][r] + sc[1][r] + sc[2][r] + sc[3][r];
      s += __shfl_xor(s, 1, 32);
      s += __shfl_xor(s, 2, 32);
      s += __shfl_xor(s, 4, 32);
      s += __shfl_xor(s, 8, 32);
      l_i[r] += s;
    }
    __syncthreads();
    // ---- ctx += P V ----
    v16bf pf0 = load_afrag_bf16(&p_lds[wave][ln][0], 0, hi);
    v16bf pf1 = load_afrag_bf16(&p_lds[wave][ln][0], 32, hi);
#pragma unroll
    for (int ht = 0; ht < 4; ++ht) {
      const __bf16* vrow = Vb + (size_t)(ht * 16 + ln) * Sd + kb;
      v16bf b0 = load_bfrag_bf16(vrow, hi);       // keys kb+0..31
      v16bf b1 = load_bfrag_bf16(vrow + 32, hi);  // keys kb+32..63
      ctx[ht] = WMMA_BF16(pf0, b0, ctx[ht]);
      ctx[ht] = WMMA_BF16(pf1, b1, ctx[ht]);
    }
  }

  // ---- normalize and write ctx merged back to [B,S,D] bf16 ----
#pragma unroll
  for (int ht = 0; ht < 4; ++ht) {
#pragma unroll
    for (int r = 0; r < 8; ++r) {
      const int s = qTile * 64 + wave * 16 + r + (hi ? 8 : 0);
      const int col = h * HDd + ht * 16 + ln;
      const float v = ctx[ht][r] / l_i[r];
      ctx_out[((size_t)b * Sd + s) * Dd + col] = (__bf16)v;
    }
  }
}

// ---------------------------------------------------------------------------
extern "C" void kernel_launch(void* const* d_in, const int* in_sizes, int n_in,
                              void* d_out, int out_size, void* d_ws, size_t ws_size,
                              hipStream_t stream) {
  const float* X    = (const float*)d_in[0];
  const float* mask = (const float*)d_in[1];
  const float* Wq = (const float*)d_in[2];
  const float* bq = (const float*)d_in[3];
  const float* Wk = (const float*)d_in[4];
  const float* bk = (const float*)d_in[5];
  const float* Wv = (const float*)d_in[6];
  const float* bv = (const float*)d_in[7];
  const float* Wo = (const float*)d_in[8];
  const float* bo = (const float*)d_in[9];
  float* out = (float*)d_out;

  const size_t MD = (size_t)Md * Dd;  // 8M elements
  const size_t DD = (size_t)Dd * Dd;  // 1M elements
  char* ws = (char*)d_ws;
  __bf16* Xb  = (__bf16*)(ws);                       // 16 MB
  __bf16* Wtq = (__bf16*)(ws + MD * 2);              // 2 MB each
  __bf16* Wtk = Wtq + DD;
  __bf16* Wtv = Wtk + DD;
  __bf16* Wto = Wtv + DD;
  __bf16* q   = Wto + DD;                            // 16 MB each
  __bf16* k   = q + MD;
  __bf16* v   = k + MD;
  __bf16* ctx = v + MD;

  // --- prep: one-time fp32->bf16 conversion / weight transposition ---
  convert_bf16<<<MD / (256 * 8), 256, 0, stream>>>(X, Xb);
  dim3 tg(Dd / 32, Dd / 32);
  transpose_cvt<<<tg, 256, 0, stream>>>(Wq, Wtq, Dd, Dd);
  transpose_cvt<<<tg, 256, 0, stream>>>(Wk, Wtk, Dd, Dd);
  transpose_cvt<<<tg, 256, 0, stream>>>(Wv, Wtv, Dd, Dd);
  transpose_cvt<<<tg, 256, 0, stream>>>(Wo, Wto, Dd, Dd);

  // --- projections (bf16 WMMA, f32 accumulate, TDM-staged A tiles) ---
  dim3 gg(Md / 128, Dd / 256);  // 64 x 4 blocks
  gemm_wmma<0><<<gg, 256, 0, stream>>>(Xb, Wtq, bq, q, Dd, Dd);
  gemm_wmma<0><<<gg, 256, 0, stream>>>(Xb, Wtk, bk, k, Dd, Dd);
  gemm_wmma<1><<<gg, 256, 0, stream>>>(Xb, Wtv, bv, v, Dd, Dd);

  // --- attention ---
  flash_attn<<<dim3(Sd / 64, Hd, Bd), 128, 0, stream>>>(q, k, v, mask, ctx);

  // --- output projection ---
  gemm_wmma<2><<<gg, 256, 0, stream>>>(ctx, Wto, bo, out, Dd, Dd);
}
